// Attention_87857851006980
// MI455X (gfx1250) — matrix operand order repaired and checked by reading, not compiled, not run on hardware
//
#include <hip/hip_runtime.h>

// ---------------- types ----------------
typedef __attribute__((ext_vector_type(16))) __bf16          v16bf;
typedef __attribute__((ext_vector_type(8)))  float           v8f;
typedef __attribute__((ext_vector_type(16))) unsigned short  us16;

#define BB   4
#define DIM  1024
#define NN   2304
#define HEADS 8
#define HD   128
#define PS   48
#define SCALE 0.03125f   // DIM^-0.5

#if defined(__HIP_DEVICE_COMPILE__) && __has_builtin(__builtin_amdgcn_global_load_async_to_lds_b128)
#define HAVE_ASYNC_LDS 1
#else
#define HAVE_ASYNC_LDS 0
#endif

// ---------------- helpers ----------------
__device__ __forceinline__ unsigned short f2bf(float f) {
    unsigned u = __builtin_bit_cast(unsigned, f);
    unsigned r = (u + 0x7FFFu + ((u >> 16) & 1u)) >> 16;
    return (unsigned short)r;
}
__device__ __forceinline__ float bf2f(unsigned short h) {
    unsigned u = ((unsigned)h) << 16;
    return __builtin_bit_cast(float, u);
}
__device__ __forceinline__ float gelu_exact(float z) {
    return 0.5f * z * (1.0f + erff(z * 0.70710678118654752f));
}
__device__ __forceinline__ v8f zero8() {
    v8f z;
#pragma unroll
    for (int i = 0; i < 8; ++i) z[i] = 0.0f;
    return z;
}
__device__ __forceinline__ v8f wmma_bf16(v16bf a, v16bf b, v8f c) {
    return __builtin_amdgcn_wmma_f32_16x16x32_bf16(false, a, false, b, (short)0, c, false, false);
}

// copy 32 bytes (16 bf16) global -> LDS; async DMA (ASYNCcnt) when available
__device__ __forceinline__ void stage32B(const unsigned short* g, unsigned short* l) {
#if HAVE_ASYNC_LDS
    typedef int v4i __attribute__((vector_size(16)));
    typedef __attribute__((address_space(1))) v4i* gp_t;
    typedef __attribute__((address_space(3))) v4i* lp_t;
    v4i* g0 = (v4i*)g;
    v4i* l0 = (v4i*)l;
    __builtin_amdgcn_global_load_async_to_lds_b128((gp_t)(g0),     (lp_t)(l0),     0, 0);
    __builtin_amdgcn_global_load_async_to_lds_b128((gp_t)(g0 + 1), (lp_t)(l0 + 1), 0, 0);
#else
    const uint4* s = (const uint4*)g;
    uint4 v0 = s[0], v1 = s[1];
    unsigned int* d = (unsigned int*)l;
    d[0]=v0.x; d[1]=v0.y; d[2]=v0.z; d[3]=v0.w; d[4]=v1.x; d[5]=v1.y; d[6]=v1.z; d[7]=v1.w;
#endif
}
__device__ __forceinline__ void async_wait0() {
#if HAVE_ASYNC_LDS
#if __has_builtin(__builtin_amdgcn_s_wait_asynccnt)
    __builtin_amdgcn_s_wait_asynccnt(0);
#else
    asm volatile("s_wait_asynccnt 0x0" ::: "memory");
#endif
#endif
}

// A fragment 16x32 from LDS laid out [row][k], row stride `stride` (u16 units)
__device__ __forceinline__ v16bf load_a_frag(const unsigned short* lds, int mbase, int stride) {
    int lane = threadIdx.x & 31;
    int m  = mbase + (lane & 15);
    int kh = lane >> 4;
    us16 us;
#pragma unroll
    for (int e = 0; e < 16; ++e) {
        int k = 16 * kh + e;
        us[e] = lds[m * stride + k];
    }
    return __builtin_bit_cast(v16bf, us);
}
// B fragment 32x16 from LDS laid out [col][k] (contraction contiguous)
__device__ __forceinline__ v16bf load_b_colk(const unsigned short* lds, int colbase, int kofs, int stride) {
    int lane = threadIdx.x & 31;
    int c  = colbase + (lane & 15);
    int kh = lane >> 4;
    us16 us;
#pragma unroll
    for (int e = 0; e < 16; ++e) us[e] = lds[c * stride + kofs + kh * 16 + e];
    return __builtin_bit_cast(v16bf, us);
}
// B fragment 32x16 from LDS laid out [k][n]
__device__ __forceinline__ v16bf load_b_rowk(const unsigned short* lds, int nbase, int stride) {
    int lane = threadIdx.x & 31;
    int n  = nbase + (lane & 15);
    int kh = lane >> 4;
    us16 us;
#pragma unroll
    for (int e = 0; e < 16; ++e) us[e] = lds[(kh * 16 + e) * stride + n];
    return __builtin_bit_cast(v16bf, us);
}

// ---------------- kernel 0: fp32 -> bf16 convert ----------------
__global__ void cvt_bf16_kernel(const float* __restrict__ src, unsigned short* __restrict__ dst, int n) {
    int i = blockIdx.x * blockDim.x + threadIdx.x;
    if (i < n) dst[i] = f2bf(src[i]);
}

// ---------------- kernel 1: transpose x(b,d,n) -> xt_bf16(b,n,d) ----------------
__global__ __launch_bounds__(256) void transpose_kernel(const float* __restrict__ x, unsigned short* __restrict__ xt) {
    __shared__ float tile[32][33];
    int b  = blockIdx.z;
    int n0 = blockIdx.x * 32;
    int d0 = blockIdx.y * 32;
    int tx = threadIdx.x & 31;
    int ty = threadIdx.x >> 5;  // 0..7
#pragma unroll
    for (int r = 0; r < 32; r += 8)
        tile[ty + r][tx] = x[((size_t)b * DIM + d0 + ty + r) * NN + n0 + tx];
    __syncthreads();
#pragma unroll
    for (int r = 0; r < 32; r += 8) {
        int n = n0 + ty + r;
        xt[((size_t)b * NN + n) * DIM + d0 + tx] = f2bf(tile[tx][ty + r]);
    }
}

// ---------------- kernel 2: pooled = mean_s gelu(bn(ximg)) ----------------
__global__ __launch_bounds__(256) void pool_kernel(const float* __restrict__ x,
                                                   const float* __restrict__ g, const float* __restrict__ be,
                                                   const float* __restrict__ mu, const float* __restrict__ var,
                                                   float* __restrict__ pooled) {
    int c = blockIdx.x;   // channel 0..1023
    int b = blockIdx.y;
    int tid = threadIdx.x;
    float inv = g[c] * rsqrtf(var[c] + 1e-5f);
    float beta = be[c], mean = mu[c];
    float acc = 0.0f;
    for (int s = tid; s < NN; s += 256) {
        int flat = c * NN + s;
        int nn = flat >> 10;
        int dd = flat & 1023;
        float v = x[((size_t)b * DIM + dd) * NN + nn];   // ximg is a raw reinterpret of xt
        acc += gelu_exact((v - mean) * inv + beta);
    }
    __shared__ float red[256];
    red[tid] = acc;
    __syncthreads();
    for (int o = 128; o > 0; o >>= 1) {
        if (tid < o) red[tid] += red[tid + o];
        __syncthreads();
    }
    if (tid == 0) pooled[b * DIM + c] = red[0] * (1.0f / (float)NN);
}

// ---------------- kernel 3: spe mix (grouped 1x1) ----------------
__global__ void spe_kernel(const float* __restrict__ pooled, const float* __restrict__ Wspe,
                           float* __restrict__ speh) {
    int i = blockIdx.x * blockDim.x + threadIdx.x;
    if (i >= BB * DIM) return;
    int b = i >> 10;
    int t = i & 1023;
    int grp = t >> 3, o = t & 7;
    float acc = 0.0f;
#pragma unroll
    for (int q = 0; q < 8; ++q)
        acc += pooled[b * DIM + grp * 8 + q] * Wspe[(grp * 8 + o) * 8 + q];
    speh[i] = acc;   // read later as speh[b*1024 + h*128 + dd]
}

// ---------------- kernel 4: grouped 3x3 conv + bn + gelu -> cbr bf16 (b,h,n,hd) ----------------
__global__ __launch_bounds__(256) void conv_kernel(const float* __restrict__ x, const float* __restrict__ Wl,
                                                   const float* __restrict__ g, const float* __restrict__ be,
                                                   const float* __restrict__ mu, const float* __restrict__ var,
                                                   unsigned short* __restrict__ cbr) {
    __shared__ float in_sl[8 * 26 * 48];   // 8 channels x 26 rows (halo) x 48
    __shared__ float wsl[8 * 8 * 9];
    int blk   = blockIdx.x;
    int strip = blk & 1;
    int grp   = (blk >> 1) & 127;
    int b     = blk >> 8;
    int y0 = strip * 24;
    for (int idx = threadIdx.x; idx < 576; idx += 256)
        wsl[idx] = Wl[(size_t)grp * 576 + idx];
    for (int idx = threadIdx.x; idx < 8 * 26 * 48; idx += 256) {
        int ic  = idx / (26 * 48);
        int rem = idx - ic * (26 * 48);
        int ry  = rem / 48;
        int xx  = rem - ry * 48;
        int y   = y0 - 1 + ry;
        float v = 0.0f;
        if (y >= 0 && y < PS) {
            int flat = (grp * 8 + ic) * NN + y * PS + xx;
            int nn = flat >> 10;
            int dd = flat & 1023;
            v = x[((size_t)b * DIM + dd) * NN + nn];
        }
        in_sl[idx] = v;
    }
    __syncthreads();
    for (int idx = threadIdx.x; idx < 8 * 24 * 48; idx += 256) {
        int o    = idx / (24 * 48);
        int rem  = idx - o * (24 * 48);
        int yloc = rem / 48;
        int xx   = rem - yloc * 48;
        int y    = y0 + yloc;
        float acc = 0.0f;
#pragma unroll
        for (int ic = 0; ic < 8; ++ic) {
            const float* ip = &in_sl[ic * (26 * 48) + yloc * 48];
            const float* wp = &wsl[o * 72 + ic * 9];
#pragma unroll
            for (int kh = 0; kh < 3; ++kh) {
                const float* row = ip + kh * 48;
#pragma unroll
                for (int kw = 0; kw < 3; ++kw) {
                    int xw = xx + kw - 1;
                    if (xw >= 0 && xw < 48) acc += row[xw] * wp[kh * 3 + kw];
                }
            }
        }
        int c = grp * 8 + o;
        float z = (acc - mu[c]) * (g[c] * rsqrtf(var[c] + 1e-5f)) + be[c];
        z = gelu_exact(z);
        int h = c >> 7, dd = c & 127;
        int pos = y * PS + xx;
        cbr[(((size_t)(b * HEADS + h)) * NN + pos) * HD + dd] = f2bf(z);
    }
}

// ---------------- kernel 5: QKV GEMM (bf16 WMMA, async double-buffered LDS) ----------------
__global__ __launch_bounds__(256) void qkv_gemm_kernel(const unsigned short* __restrict__ A,   // xt bf16 (9216x1024)
                                                       const unsigned short* __restrict__ Bw,  // Wqkv bf16 (3072x1024)
                                                       const float* __restrict__ bias,
                                                       unsigned short* __restrict__ qb,
                                                       unsigned short* __restrict__ kb,
                                                       unsigned short* __restrict__ vb) {
    __shared__ unsigned short a_t[2][128 * 40];
    __shared__ unsigned short b_t[2][128 * 40];
    int colblk = blockIdx.x;   // 0..23
    int rowblk = blockIdx.y;   // 0..71
    int tid = threadIdx.x;
    int w = tid >> 5, lane = tid & 31;
    int wr = w >> 1, wc = w & 1;
    v8f acc[2][4];
#pragma unroll
    for (int i = 0; i < 2; ++i)
#pragma unroll
        for (int j = 0; j < 4; ++j) acc[i][j] = zero8();
    int lrow = tid >> 1, lhalf = tid & 1;
    const unsigned short* gA = A  + ((size_t)(rowblk * 128 + lrow)) * DIM + lhalf * 16;
    const unsigned short* gB = Bw + ((size_t)(colblk * 128 + lrow)) * DIM + lhalf * 16;
    int lofs = lrow * 40 + lhalf * 16;

    stage32B(gA, &a_t[0][lofs]);
    stage32B(gB, &b_t[0][lofs]);
    for (int t = 0; t < 32; ++t) {
        int cur = t & 1;
        async_wait0();
        __syncthreads();
        if (t + 1 < 32) {
            int nxt = cur ^ 1;
            stage32B(gA + (t + 1) * 32, &a_t[nxt][lofs]);
            stage32B(gB + (t + 1) * 32, &b_t[nxt][lofs]);
        }
        const unsigned short* at = a_t[cur];
        const unsigned short* bt = b_t[cur];
        v16bf a0 = load_a_frag(at, wr * 32,      40);
        v16bf a1 = load_a_frag(at, wr * 32 + 16, 40);
        v16bf b0 = load_b_colk(bt, wc * 64 +  0, 0, 40);
        v16bf b1 = load_b_colk(bt, wc * 64 + 16, 0, 40);
        v16bf b2 = load_b_colk(bt, wc * 64 + 32, 0, 40);
        v16bf b3 = load_b_colk(bt, wc * 64 + 48, 0, 40);
        acc[0][0] = wmma_bf16(a0, b0, acc[0][0]);
        acc[0][1] = wmma_bf16(a0, b1, acc[0][1]);
        acc[0][2] = wmma_bf16(a0, b2, acc[0][2]);
        acc[0][3] = wmma_bf16(a0, b3, acc[0][3]);
        acc[1][0] = wmma_bf16(a1, b0, acc[1][0]);
        acc[1][1] = wmma_bf16(a1, b1, acc[1][1]);
        acc[1][2] = wmma_bf16(a1, b2, acc[1][2]);
        acc[1][3] = wmma_bf16(a1, b3, acc[1][3]);
        __syncthreads();
    }
    int kh = lane >> 4, ln = lane & 15;
#pragma unroll
    for (int i = 0; i < 2; ++i)
#pragma unroll
        for (int j = 0; j < 4; ++j) {
            int col = colblk * 128 + wc * 64 + j * 16 + ln;
            int sel = col >> 10;
            int jj  = col & 1023;
            int h = jj >> 7, dd = jj & 127;
            unsigned short* dst = (sel == 0) ? qb : ((sel == 1) ? kb : vb);
            float bcol = bias[col];
#pragma unroll
            for (int r = 0; r < 8; ++r) {
                int m = rowblk * 128 + wr * 32 + i * 16 + kh * 8 + r;
                int bidx = m / NN;
                int n = m - bidx * NN;
                dst[(((size_t)(bidx * HEADS + h)) * NN + n) * HD + dd] = f2bf(acc[i][j][r] + bcol);
            }
        }
}

// ---------------- kernel 6: vpc = v_sum*spe*scale + cbr ----------------
__global__ __launch_bounds__(256) void vpc_kernel(const unsigned short* __restrict__ vb,
                                                  const unsigned short* __restrict__ cbr,
                                                  const float* __restrict__ speh,
                                                  unsigned short* __restrict__ vpc) {
    int row = blockIdx.x * blockDim.x + threadIdx.x;   // (b,h,n)
    if (row >= BB * HEADS * NN) return;
    int bh = row / NN;
    int b = bh >> 3, h = bh & 7;
    size_t base = (size_t)row * HD;
    float s = 0.0f;
    for (int dd = 0; dd < HD; ++dd) s += bf2f(vb[base + dd]);
    s *= SCALE;
    const float* sp = &speh[b * DIM + h * HD];
    for (int dd = 0; dd < HD; ++dd)
        vpc[base + dd] = f2bf(s * sp[dd] + bf2f(cbr[base + dd]));
}

// ---------------- kernel 7: flash attention (bf16 WMMA, async double-buffered LDS) ----------------
__global__ __launch_bounds__(256) void attn_kernel(const unsigned short* __restrict__ qb,
                                                   const unsigned short* __restrict__ kb,
                                                   const unsigned short* __restrict__ vpc,
                                                   unsigned short* __restrict__ ob) {
    __shared__ unsigned short kt_lds[2][32 * 136];
    __shared__ unsigned short vt_lds[2][32 * 136];
    __shared__ unsigned short p_lds[8][16 * 36];
    int bh = blockIdx.y;                  // 0..31
    int b = bh >> 3, h = bh & 7;
    int w = threadIdx.x >> 5, lane = threadIdx.x & 31;
    int qbase = (blockIdx.x * 8 + w) * 16;
    size_t qoff = ((size_t)bh * NN + qbase) * HD;
    // q A-fragments gathered from global (reused across the whole key loop)
    v16bf qa[4];
    {
        int m = lane & 15, kh = lane >> 4;
#pragma unroll
        for (int kc = 0; kc < 4; ++kc) {
            us16 us;
#pragma unroll
            for (int e = 0; e < 16; ++e) {
                int k = kc * 32 + 16 * kh + e;
                us[e] = qb[qoff + (size_t)m * HD + k];
            }
            qa[kc] = __builtin_bit_cast(v16bf, us);
        }
    }
    v8f oacc[8];
#pragma unroll
    for (int i = 0; i < 8; ++i) oacc[i] = zero8();
    float mrow[8], lrow[8];
#pragma unroll
    for (int r = 0; r < 8; ++r) { mrow[r] = -1e30f; lrow[r] = 0.0f; }

    int key = threadIdx.x >> 3, seg = threadIdx.x & 7;
    const unsigned short* gK = kb  + ((size_t)bh * NN + key) * HD + seg * 16;
    const unsigned short* gV = vpc + ((size_t)bh * NN + key) * HD + seg * 16;
    int sofs = key * 136 + seg * 16;

    stage32B(gK, &kt_lds[0][sofs]);
    stage32B(gV, &vt_lds[0][sofs]);
    for (int kt = 0; kt < NN / 32; ++kt) {
        int cur = kt & 1;
        async_wait0();
        __syncthreads();
        if (kt + 1 < NN / 32) {
            int nxt = cur ^ 1;
            size_t adv = (size_t)(kt + 1) * 32 * HD;
            stage32B(gK + adv, &kt_lds[nxt][sofs]);
            stage32B(gV + adv, &vt_lds[nxt][sofs]);
        }
        const unsigned short* kl = kt_lds[cur];
        const unsigned short* vl = vt_lds[cur];
        // S = q k^T * scale  (16 x 32)
        v8f s0 = zero8(), s1 = zero8();
#pragma unroll
        for (int kc = 0; kc < 4; ++kc) {
            v16bf kf0 = load_b_colk(kl,  0, kc * 32, 136);
            v16bf kf1 = load_b_colk(kl, 16, kc * 32, 136);
            s0 = wmma_bf16(qa[kc], kf0, s0);
            s1 = wmma_bf16(qa[kc], kf1, s1);
        }
#pragma unroll
        for (int r = 0; r < 8; ++r) { s0[r] *= SCALE; s1[r] *= SCALE; }
        float rmax[8];
#pragma unroll
        for (int r = 0; r < 8; ++r) rmax[r] = fmaxf(s0[r], s1[r]);
#pragma unroll
        for (int off = 1; off <= 8; off <<= 1)
#pragma unroll
            for (int r = 0; r < 8; ++r) rmax[r] = fmaxf(rmax[r], __shfl_xor(rmax[r], off, 32));
        float mnew[8], alpha[8];
#pragma unroll
        for (int r = 0; r < 8; ++r) {
            mnew[r]  = fmaxf(mrow[r], rmax[r]);
            alpha[r] = expf(mrow[r] - mnew[r]);
        }
        v8f p0, p1;
#pragma unroll
        for (int r = 0; r < 8; ++r) {
            p0[r] = expf(s0[r] - mnew[r]);
            p1[r] = expf(s1[r] - mnew[r]);
        }
        float rs[8];
#pragma unroll
        for (int r = 0; r < 8; ++r) rs[r] = p0[r] + p1[r];
#pragma unroll
        for (int off = 1; off <= 8; off <<= 1)
#pragma unroll
            for (int r = 0; r < 8; ++r) rs[r] += __shfl_xor(rs[r], off, 32);
#pragma unroll
        for (int r = 0; r < 8; ++r) {
            lrow[r] = lrow[r] * alpha[r] + rs[r];
            mrow[r] = mnew[r];
        }
#pragma unroll
        for (int ct = 0; ct < 8; ++ct)
#pragma unroll
            for (int r = 0; r < 8; ++r) oacc[ct][r] *= alpha[r];
        // P (C layout) -> per-wave LDS scratch -> A fragment layout
        {
            unsigned short* pl = p_lds[w];
            int colb = lane & 15, kh = lane >> 4;
#pragma unroll
            for (int r = 0; r < 8; ++r) {
                int row = kh * 8 + r;
                pl[row * 36 + colb]      = f2bf(p0[r]);
                pl[row * 36 + 16 + colb] = f2bf(p1[r]);
            }
        }
        v16bf pa = load_a_frag(p_lds[w], 0, 36);
#pragma unroll
        for (int ct = 0; ct < 8; ++ct) {
            v16bf vf = load_b_rowk(vl, ct * 16, 136);
            oacc[ct] = wmma_bf16(pa, vf, oacc[ct]);
        }
        __syncthreads();
    }
    // epilogue: O / l  -> (b, n, d) bf16
    int kh = lane >> 4, ln = lane & 15;
#pragma unroll
    for (int ct = 0; ct < 8; ++ct)
#pragma unroll
        for (int r = 0; r < 8; ++r) {
            float val = oacc[ct][r] / lrow[r];
            int row = qbase + kh * 8 + r;
            ob[((size_t)(b * NN + row)) * DIM + h * HD + ct * 16 + ln] = f2bf(val);
        }
}

// ---------------- kernel 8: output GEMM -> d_out fp32 ----------------
__global__ __launch_bounds__(256) void out_gemm_kernel(const unsigned short* __restrict__ A,   // o bf16 (9216x1024)
                                                       const unsigned short* __restrict__ Bw,  // Wnn1 bf16 (1024x1024)
                                                       const float* __restrict__ bias,
                                                       float* __restrict__ out) {
    __shared__ unsigned short a_t[2][128 * 40];
    __shared__ unsigned short b_t[2][128 * 40];
    int colblk = blockIdx.x;   // 0..7
    int rowblk = blockIdx.y;   // 0..71
    int tid = threadIdx.x;
    int w = tid >> 5, lane = tid & 31;
    int wr = w >> 1, wc = w & 1;
    v8f acc[2][4];
#pragma unroll
    for (int i = 0; i < 2; ++i)
#pragma unroll
        for (int j = 0; j < 4; ++j) acc[i][j] = zero8();
    int lrow = tid >> 1, lhalf = tid & 1;
    const unsigned short* gA = A  + ((size_t)(rowblk * 128 + lrow)) * DIM + lhalf * 16;
    const unsigned short* gB = Bw + ((size_t)(colblk * 128 + lrow)) * DIM + lhalf * 16;
    int lofs = lrow * 40 + lhalf * 16;

    stage32B(gA, &a_t[0][lofs]);
    stage32B(gB, &b_t[0][lofs]);
    for (int t = 0; t < 32; ++t) {
        int cur = t & 1;
        async_wait0();
        __syncthreads();
        if (t + 1 < 32) {
            int nxt = cur ^ 1;
            stage32B(gA + (t + 1) * 32, &a_t[nxt][lofs]);
            stage32B(gB + (t + 1) * 32, &b_t[nxt][lofs]);
        }
        const unsigned short* at = a_t[cur];
        const unsigned short* bt = b_t[cur];
        v16bf a0 = load_a_frag(at, wr * 32,      40);
        v16bf a1 = load_a_frag(at, wr * 32 + 16, 40);
        v16bf b0 = load_b_colk(bt, wc * 64 +  0, 0, 40);
        v16bf b1 = load_b_colk(bt, wc * 64 + 16, 0, 40);
        v16bf b2 = load_b_colk(bt, wc * 64 + 32, 0, 40);
        v16bf b3 = load_b_colk(bt, wc * 64 + 48, 0, 40);
        acc[0][0] = wmma_bf16(a0, b0, acc[0][0]);
        acc[0][1] = wmma_bf16(a0, b1, acc[0][1]);
        acc[0][2] = wmma_bf16(a0, b2, acc[0][2]);
        acc[0][3] = wmma_bf16(a0, b3, acc[0][3]);
        acc[1][0] = wmma_bf16(a1, b0, acc[1][0]);
        acc[1][1] = wmma_bf16(a1, b1, acc[1][1]);
        acc[1][2] = wmma_bf16(a1, b2, acc[1][2]);
        acc[1][3] = wmma_bf16(a1, b3, acc[1][3]);
        __syncthreads();
    }
    int kh = lane >> 4, ln = lane & 15;
#pragma unroll
    for (int i = 0; i < 2; ++i)
#pragma unroll
        for (int j = 0; j < 4; ++j) {
            int col = colblk * 128 + wc * 64 + j * 16 + ln;
            float bcol = bias[col];
#pragma unroll
            for (int r = 0; r < 8; ++r) {
                int m = rowblk * 128 + wr * 32 + i * 16 + kh * 8 + r;
                out[(size_t)m * DIM + col] = acc[i][j][r] + bcol;
            }
        }
}

// ---------------- host launcher ----------------
extern "C" void kernel_launch(void* const* d_in, const int* in_sizes, int n_in,
                              void* d_out, int out_size, void* d_ws, size_t ws_size,
                              hipStream_t stream) {
    const float* x      = (const float*)d_in[0];
    const float* Wqkv   = (const float*)d_in[1];
    const float* bqkv   = (const float*)d_in[2];
    const float* Wspe   = (const float*)d_in[3];
    const float* Wlocal = (const float*)d_in[4];
    const float* Wnn1   = (const float*)d_in[5];
    const float* bnn1   = (const float*)d_in[6];
    const float* bn_g   = (const float*)d_in[7];
    const float* bn_b   = (const float*)d_in[8];
    const float* bn_m   = (const float*)d_in[9];
    const float* bn_v   = (const float*)d_in[10];
    const float* bnc_g  = (const float*)d_in[11];
    const float* bnc_b  = (const float*)d_in[12];
    const float* bnc_m  = (const float*)d_in[13];
    const float* bnc_v  = (const float*)d_in[14];
    float* out = (float*)d_out;

    char* ws = (char*)d_ws;
    size_t off = 0;
    auto alloc = [&](size_t bytes) -> void* {
        void* p = ws + off;
        off += (bytes + 255) & ~(size_t)255;
        return p;
    };
    const size_t NE = (size_t)BB * NN * DIM;   // 9,437,184 elements
    unsigned short* xt   = (unsigned short*)alloc(NE * 2);   // xt bf16; later aliased as attn output o
    unsigned short* qb   = (unsigned short*)alloc(NE * 2);
    unsigned short* kbuf = (unsigned short*)alloc(NE * 2);
    unsigned short* vbuf = (unsigned short*)alloc(NE * 2);
    unsigned short* vpc  = (unsigned short*)alloc(NE * 2);
    unsigned short* cbr  = (unsigned short*)alloc(NE * 2);
    unsigned short* wq   = (unsigned short*)alloc((size_t)3 * DIM * DIM * 2);
    unsigned short* w1   = (unsigned short*)alloc((size_t)DIM * DIM * 2);
    float* pooled        = (float*)alloc((size_t)BB * DIM * 4);
    float* speh          = (float*)alloc((size_t)BB * DIM * 4);

    cvt_bf16_kernel<<<(3 * DIM * DIM + 255) / 256, 256, 0, stream>>>(Wqkv, wq, 3 * DIM * DIM);
    cvt_bf16_kernel<<<(DIM * DIM + 255) / 256, 256, 0, stream>>>(Wnn1, w1, DIM * DIM);
    transpose_kernel<<<dim3(NN / 32, DIM / 32, BB), 256, 0, stream>>>(x, xt);
    pool_kernel<<<dim3(DIM, BB), 256, 0, stream>>>(x, bn_g, bn_b, bn_m, bn_v, pooled);
    spe_kernel<<<(BB * DIM + 255) / 256, 256, 0, stream>>>(pooled, Wspe, speh);
    conv_kernel<<<BB * 128 * 2, 256, 0, stream>>>(x, Wlocal, bnc_g, bnc_b, bnc_m, bnc_v, cbr);
    qkv_gemm_kernel<<<dim3(24, 72), 256, 0, stream>>>(xt, wq, bqkv, qb, kbuf, vbuf);
    vpc_kernel<<<(BB * HEADS * NN + 255) / 256, 256, 0, stream>>>(vbuf, cbr, speh, vpc);
    attn_kernel<<<dim3(NN / (16 * 8), BB * HEADS), 256, 0, stream>>>(qb, kbuf, vpc, xt);
    out_gemm_kernel<<<dim3(8, 72), 256, 0, stream>>>(xt, w1, bnn1, out);
    (void)in_sizes; (void)n_in; (void)out_size; (void)ws_size;
}
